// FilterStripe_31009663877374
// MI455X (gfx1250) — matrix operand order repaired and checked by reading, not compiled, not run on hardware
//
#include <hip/hip_runtime.h>

// ---------------- problem constants ----------------
#define CIN    128
#define COUT   128
#define HH     56
#define WW     56
#define HW     (HH * WW)          // 3136
#define NBATCH 32
#define TAPS   9                  // 3x3 kernel taps
// ---------------- tiling constants -----------------
#define PT     32                 // flattened output positions per block
#define TILES  (HW / PT)          // 98 tiles per image
#define ROWS   4                  // staged input rows (2 output rows + halo)
#define COLS   58                 // 56 + left/right zero halo
#define XPAD   40                 // c-stride (shorts): 80B -> 16B aligned, 20-dword bank stride
#define CO_STEP 32                // input channels per slice
#define XS_SHORTS (ROWS * COLS * XPAD)       // 9,280  (18,560 B)
#define WL_SHORTS (TAPS * COUT * 32)         // 36,864 (73,728 B)
#define WL_LDS_OFF (XS_SHORTS * 2)           // byte offset of wl inside LDS block

typedef __attribute__((ext_vector_type(16))) __bf16        v16bf;
typedef __attribute__((ext_vector_type(8)))  float         v8f;
typedef __attribute__((ext_vector_type(4)))  unsigned int  v4u;
typedef __attribute__((ext_vector_type(8)))  int           v8i;
typedef __attribute__((ext_vector_type(4)))  int           v4i;

#define TDM_OK (__has_builtin(__builtin_amdgcn_tensor_load_to_lds) && \
                __has_builtin(__builtin_amdgcn_s_wait_tensorcnt))

__device__ __forceinline__ unsigned short f2bf(float f) {
  union { float f; unsigned int u; } v; v.f = f;
  unsigned int u = v.u;
  u += 0x7fffu + ((u >> 16) & 1u);   // round-to-nearest-even
  return (unsigned short)(u >> 16);
}

union FragBF {
  uint4 q[2];   // 32 bytes = 16 bf16
  v16bf v;
};

struct __align__(16) SMem {
  unsigned short xs[XS_SHORTS];   // input window bf16 [row][col][c]
  unsigned short wl[WL_SHORTS];   // weights bf16     [tap][o][c32]
};

#if TDM_OK
// One 2D TDM: 9 tap-rows of 2048 dwords (8KB) each, global row stride 8192 dwords,
// packed contiguously into LDS at lds_off.  Dims/strides in data_size (4B) units.
__device__ __forceinline__ void tdm_load_weights(const unsigned short* gsrc,
                                                 unsigned int lds_off) {
  unsigned long long ga = (unsigned long long)gsrc;
  v4u g0;
  g0[0] = 1u;                                              // D# count = 1
  g0[1] = lds_off;                                         // lds_addr
  g0[2] = (unsigned int)ga;                                // global_addr[31:0]
  g0[3] = (unsigned int)((ga >> 32) & 0x01ffffffu)         // global_addr[56:32]
        | (2u << 30);                                      // type = 2 (image)
  v8i g1;
  g1[0] = (int)(2u << 16);       // data_size = 4B
  g1[1] = (int)(2048u << 16);    // tensor_dim0 = 2048 (lo16 at [63:48])
  g1[2] = (int)(9u << 16);       // tensor_dim0 hi16 = 0 | tensor_dim1 = 9
  g1[3] = (int)(2048u << 16);    // tensor_dim1 hi16 = 0 | tile_dim0 = 2048
  g1[4] = 9;                     // tile_dim1 = 9 | tile_dim2 = 0
  g1[5] = 8192;                  // tensor_dim0_stride = 8192 dwords (4 chunks)
  g1[6] = 0;
  g1[7] = 0;
  v4i z4 = {0, 0, 0, 0};
#if __clang_major__ >= 23
  v8i z8 = {0, 0, 0, 0, 0, 0, 0, 0};
  __builtin_amdgcn_tensor_load_to_lds(g0, g1, z4, z4, z8, 0);
#else
  __builtin_amdgcn_tensor_load_to_lds(g0, g1, z4, z4, 0);
#endif
}
#endif

// Convert weight f32 [tap*COUT + o][c] -> bf16 chunk-major [tap][c/32][o][c%32]
__global__ __launch_bounds__(256)
void wprep_bf16(const float* __restrict__ wsrc, unsigned short* __restrict__ wdst) {
  int idx = blockIdx.x * 256 + threadIdx.x;
  if (idx >= TAPS * COUT * CIN) return;
  int c   = idx % CIN;
  int row = idx / CIN;              // tap*COUT + o
  int tap = row / COUT;
  int o   = row % COUT;
  int q   = c >> 5;
  int cl  = c & 31;
  wdst[(((tap << 2) + q) * COUT + o) * 32 + cl] = f2bf(wsrc[idx]);
}

__global__ __launch_bounds__(256)
void conv3x3_wmma_bf16(const float* __restrict__ x,
                       const unsigned short* __restrict__ wbf,
                       float* __restrict__ out) {
  __shared__ SMem sm;               // 92,288 B (WGP LDS pool is 320 KB)

  const int tid   = threadIdx.x;
  const int lane  = tid & 31;
  const int wave  = tid >> 5;       // 8 waves
  const int ln16  = lane & 15;
  const int hi    = (lane >> 4) & 1;

  const int tile  = blockIdx.x % TILES;
  const int n     = blockIdx.x / TILES;
  const int p0    = tile * PT;
  const int h0    = p0 / WW;
  const int hbase = h0 - 1;

  // Per-lane B-matrix positions (col N = ln16 within each 16-wide half)
  int rb[2], wb[2];
#pragma unroll
  for (int j = 0; j < 2; ++j) {
    int p = p0 + j * 16 + ln16;
    rb[j] = p / WW - h0;            // 0 or 1
    wb[j] = p % WW;
  }

  const int o0 = wave * 16;

  v8f acc[2];
#pragma unroll
  for (int j = 0; j < 2; ++j)
#pragma unroll
    for (int r = 0; r < 8; ++r) acc[j][r] = 0.0f;

  // ---- zero the input window ONCE: covers the constant halo (w=-1, w=56,
  //      out-of-range h rows) which is never rewritten by slice staging ----
  {
    unsigned int* z = (unsigned int*)sm.xs;
    for (int i = tid; i < XS_SHORTS / 2; i += 256) z[i] = 0u;
  }

  for (int co = 0; co < CIN; co += CO_STEP) {
    __syncthreads();                // previous slice fully consumed (also fences zero-fill)
    const int q = co >> 5;

#if TDM_OK
    if (wave == 0)
      tdm_load_weights(wbf + q * 4096, WL_LDS_OFF);   // async: 73,728 B -> LDS
#else
    {
      const unsigned int* srcb = (const unsigned int*)wbf + q * 2048;
      unsigned int* dst = (unsigned int*)sm.wl;
      for (int i = tid; i < TAPS * 2048; i += 256) {
        int tap = i >> 11, e = i & 2047;
        dst[i] = srcb[tap * 8192 + e];
      }
    }
#endif

    // ---- stage interior of input window, vectorized 4 floats (one uint4) per step:
    //      rows h in [h0-1, h0+2] (skip OOB: those LDS rows stay zero),
    //      w in [0,56) as 14 quads, channels [co, co+32) ----
    for (int i = tid; i < ROWS * CO_STEP * 14; i += 256) {
      int quad = i % 14;
      int t    = i / 14;
      int cl   = t & 31;
      int r    = t >> 5;
      int h    = hbase + r;
      if ((unsigned)h < HH) {
        int w0 = quad * 4;
        const uint4* src = (const uint4*)(x + (n * CIN + co + cl) * HW + h * WW + w0);
        uint4 d = *src;
        union { uint4 u; float f[4]; } v; v.u = d;
        unsigned short* dst = sm.xs + (r * COLS + (w0 + 1)) * XPAD + cl;
#pragma unroll
        for (int k = 0; k < 4; ++k) dst[k * XPAD] = f2bf(v.f[k]);
      }
    }

#if TDM_OK
    if (wave == 0) __builtin_amdgcn_s_wait_tensorcnt(0);
#endif
    __syncthreads();                // weights (TDM) + inputs visible to all waves

    // ---- 9 taps, no inner barriers: 18 WMMAs per wave per slice ----
#pragma unroll
    for (int kh = 0; kh < 3; ++kh) {
#pragma unroll
      for (int kw = 0; kw < 3; ++kw) {
        const int tap = kh * 3 + kw;

        // A fragment: 16x32 bf16 (ISA 7.12.2 layout)
        FragBF A;
        {
          int klo = hi ? 8 : 0;
          const unsigned short* ap = sm.wl + tap * 4096 + (o0 + ln16) * 32 + klo;
          A.q[0] = *(const uint4*)(ap);
          A.q[1] = *(const uint4*)(ap + 16);
        }

#pragma unroll
        for (int j = 0; j < 2; ++j) {
          // B fragment: 32x16 bf16, K contiguous in channels
          FragBF B;
          const unsigned short* bp =
              sm.xs + ((rb[j] + kh) * COLS + (wb[j] + kw)) * XPAD + hi * 8;
          B.q[0] = *(const uint4*)(bp);
          B.q[1] = *(const uint4*)(bp + 16);

          acc[j] = __builtin_amdgcn_wmma_f32_16x16x32_bf16(
              false, A.v, false, B.v, (short)0, acc[j], false, false);
        }
      }
    }
  }

  // ---- store D: VGPR r -> (M = r + 8*hi, N = lane%16) ----
#pragma unroll
  for (int j = 0; j < 2; ++j) {
    int p = p0 + j * 16 + ln16;
#pragma unroll
    for (int r = 0; r < 8; ++r) {
      int o = o0 + r + hi * 8;
      out[(n * COUT + o) * HW + p] = acc[j][r];
    }
  }
}

extern "C" void kernel_launch(void* const* d_in, const int* in_sizes, int n_in,
                              void* d_out, int out_size, void* d_ws, size_t ws_size,
                              hipStream_t stream) {
  const float* x = (const float*)d_in[0];            // (32,128,56,56) f32
  const float* w = (const float*)d_in[1];            // (1152,128) f32
  float* out = (float*)d_out;                        // (32,128,56,56) f32
  unsigned short* wbf = (unsigned short*)d_ws;       // 294,912 B bf16 weights

  const int welems = TAPS * COUT * CIN;              // 147,456
  wprep_bf16<<<(welems + 255) / 256, 256, 0, stream>>>(w, wbf);
  conv3x3_wmma_bf16<<<NBATCH * TILES, 256, 0, stream>>>(x, wbf, out);
}
